// DecoderLayer_40372692583270
// MI455X (gfx1250) — matrix operand order repaired and checked by reading, not compiled, not run on hardware
//
#include <hip/hip_runtime.h>

typedef _Float16 f16;
typedef __attribute__((ext_vector_type(16))) _Float16 v16h;
typedef __attribute__((ext_vector_type(8)))  float    v8f;

#define DEVI static __device__ __forceinline__

// ---- WMMA operand loaders (CDNA5 16-bit 16x16x32 layouts, ISA 7.12.2) ----
// A (16x32, MxK): lanes 0-15 row M=lane, K = {0..7, 16..23}; lanes 16-31 same M,
// K = {8..15, 24..31}. 16 halves per lane = two 16-byte chunks at +kb and +kb+16.
DEVI v16h loadA16(const f16* p, int kb) {
  union { v16h v; float4 q[2]; } u;
  u.q[0] = *(const float4*)(p + kb);
  u.q[1] = *(const float4*)(p + kb + 16);
  return u.v;
}
// B (32x16, KxN) from B^T rows (column-contiguous): lanes 0-15 col N=lane hold
// K = {0..7, 16..23}; lanes 16-31 hold K = {8..15, 24..31}. Chunks at +kb, +kb+16.
DEVI v16h loadB16(const f16* p, int kb) {
  union { v16h v; float4 q[2]; } u;
  u.q[0] = *(const float4*)(p + kb);
  u.q[1] = *(const float4*)(p + kb + 16);
  return u.v;
}

// ---------------------------------------------------------------------------
// Generic tap-fused WMMA GEMM, register-blocked 32(M) x 64(N) per wave:
//   out[n, col, co] = epilogue( sum_tap sum_k A[tap][m, k] * BT[n, j(col,tap), k] )
// mode 0: j = col + tap - (ntaps==3 ? 1 : 0)      (conv pad=1 / plain GEMM)
// mode 1: j = (col + 1 - tap)/2 if even & <Lin    (ConvTranspose1d k3 s2 p1 op1)
// Invalid columns read from `zrow` (pre-zeroed K-length row) -> no inner-loop
// masking; EXEC stays all-ones for every WMMA.
// epilogue: per-channel scale/shift (sc,sh) or uniform scale; +resid; ReLU;
//           writes f32 and/or f16, both contiguous along co.
// Requires M % 32 == 0, Lout % 64 == 0, K % 32 == 0.
// ---------------------------------------------------------------------------
__global__ __launch_bounds__(128) void k_gemm(
    const f16* __restrict__ A, int lda, long long sAn,
    const f16* __restrict__ BT, int ldb, long long sBn,
    const f16* __restrict__ zrow,
    float* __restrict__ outF, f16* __restrict__ outH,
    int ldo, long long sOn, int co_base,
    const float* __restrict__ sc, const float* __restrict__ sh,
    float uscale, const float* __restrict__ resid, long long sRn,
    int M, int K, int Lout, int Lin, int nBatch, int ntaps, int mode, int relu)
{
  const int wid = blockIdx.x * (blockDim.x >> 5) + (threadIdx.x >> 5);
  const int mt = M >> 5;     // 32-row blocks
  const int lt = Lout >> 6;  // 64-col blocks
  const int per = mt * lt;
  if (wid >= per * nBatch) return;
  const int n = wid / per;
  const int rr = wid - n * per;
  const int mtile = rr % mt;
  const int ltile = rr / mt;
  const int lane = threadIdx.x & 31;
  const int nl = lane & 15;
  const int mbase = mtile << 5;
  const int cbase = ltile << 6;
  const int kbA = (lane < 16) ? 0 : 8;
  const int kbB = (lane < 16) ? 0 : 8;

  int cols[4];
#pragma unroll
  for (int ci = 0; ci < 4; ++ci) cols[ci] = cbase + (ci << 4) + nl;

  const v8f vzero = {0.f, 0.f, 0.f, 0.f, 0.f, 0.f, 0.f, 0.f};
  v8f acc[2][4];
#pragma unroll
  for (int mi = 0; mi < 2; ++mi)
#pragma unroll
    for (int ci = 0; ci < 4; ++ci) acc[mi][ci] = vzero;

  const f16* An0 = A + (long long)n * sAn + (long long)(mbase + nl) * lda;
  const f16* An1 = An0 + (long long)16 * lda;
  const f16* Bn = BT + (long long)n * sBn;

  for (int tap = 0; tap < ntaps; ++tap) {
    const f16* Arow0 = An0 + (long long)tap * M * lda;
    const f16* Arow1 = An1 + (long long)tap * M * lda;
    const f16* Brow[4];
#pragma unroll
    for (int ci = 0; ci < 4; ++ci) {
      int j; bool valid;
      if (mode == 1) {
        int num = cols[ci] + 1 - tap;
        j = num >> 1;
        valid = (num >= 0) && ((num & 1) == 0) && (j < Lin);
      } else {
        j = cols[ci] + tap - ((ntaps == 3) ? 1 : 0);
        valid = (j >= 0) && (j < Lin);
      }
      Brow[ci] = valid ? (Bn + (long long)j * ldb) : zrow;  // hoisted select
    }
    for (int kc = 0; kc < K; kc += 32) {
      v16h a0 = loadA16(Arow0 + kc, kbA);
      v16h a1 = loadA16(Arow1 + kc, kbA);
      v16h b[4];
#pragma unroll
      for (int ci = 0; ci < 4; ++ci) b[ci] = loadB16(Brow[ci] + kc, kbB);
#pragma unroll
      for (int ci = 0; ci < 4; ++ci) {
        acc[0][ci] = __builtin_amdgcn_wmma_f32_16x16x32_f16(
            false, a0, false, b[ci], (short)0, acc[0][ci], false, false);
        acc[1][ci] = __builtin_amdgcn_wmma_f32_16x16x32_f16(
            false, a1, false, b[ci], (short)0, acc[1][ci], false, false);
      }
    }
  }

  // Epilogue. C/D tile: VGPR r holds (M = r + (lane<16?0:8), N = lane&15).
  const int r0 = (lane < 16) ? 0 : 8;
#pragma unroll
  for (int mi = 0; mi < 2; ++mi) {
    const int cloc = mbase + (mi << 4) + r0;  // channel base (for sc/sh)
    const int cb = co_base + cloc;            // memory channel base
#pragma unroll
    for (int ci = 0; ci < 4; ++ci) {
      const int col = cols[ci];
      const long long obase = (long long)n * sOn + (long long)col * ldo + cb;
      float vals[8];
#pragma unroll
      for (int i = 0; i < 8; ++i) {
        float v = acc[mi][ci][i];
        if (sc) v = v * sc[cloc + i] + sh[cloc + i];
        else    v = v * uscale;
        if (resid) v += resid[(long long)n * sRn + (long long)col * ldo + cb + i];
        if (relu) v = v > 0.f ? v : 0.f;
        vals[i] = v;
      }
      if (outF) {
        float4* o = (float4*)(outF + obase);
        o[0] = make_float4(vals[0], vals[1], vals[2], vals[3]);
        o[1] = make_float4(vals[4], vals[5], vals[6], vals[7]);
      }
      if (outH) {
        union { float4 q; f16 h[8]; } u;
#pragma unroll
        for (int i = 0; i < 8; ++i) u.h[i] = (f16)vals[i];
        *(float4*)(outH + obase) = u.q;
      }
    }
  }
}

// ---- small helper kernels ------------------------------------------------

__global__ void k_zero_h(f16* __restrict__ p, int count) {
  int i = blockIdx.x * blockDim.x + threadIdx.x;
  if (i < count) p[i] = (f16)0.f;
}

// Fold conv-bias + BN into per-channel scale/shift: y = x*s + t
__global__ void k_bnprep(const float* __restrict__ g, const float* __restrict__ be,
                         const float* __restrict__ rm, const float* __restrict__ rv,
                         const float* __restrict__ bias,
                         float* __restrict__ s, float* __restrict__ t, int c) {
  int i = blockIdx.x * blockDim.x + threadIdx.x;
  if (i >= c) return;
  float sv = g[i] * rsqrtf(rv[i] + 1e-5f);
  s[i] = sv;
  float bb = bias ? bias[i] : 0.f;
  t[i] = (bb - rm[i]) * sv + be[i];
}

__global__ void k_upprep(const float* __restrict__ bias,
                         float* __restrict__ s, float* __restrict__ t, int c) {
  int i = blockIdx.x * blockDim.x + threadIdx.x;
  if (i >= c) return;
  s[i] = 1.f;
  t[i] = bias[i];
}

// conv weight -> f16 staged as [tap][out][in]; src elem = w[o*so + i*si + k]
__global__ void k_cvt_w3(const float* __restrict__ w, f16* __restrict__ o,
                         int O, int I, int so, int si) {
  long long idx = (long long)blockIdx.x * blockDim.x + threadIdx.x;
  long long total = 3LL * O * I;
  if (idx >= total) return;
  int k = (int)(idx / ((long long)O * I));
  long long rem = idx - (long long)k * O * I;
  int oo = (int)(rem / I);
  int ii = (int)(rem - (long long)oo * I);
  o[idx] = (f16)w[(long long)oo * so + (long long)ii * si + k];
}

__global__ void k_cvt_mat(const float* __restrict__ w, f16* __restrict__ o,
                          long long total) {
  long long i = (long long)blockIdx.x * blockDim.x + threadIdx.x;
  if (i >= total) return;
  o[i] = (f16)w[i];
}

// (N,C,L) f32 -> (N,L,C) f16
__global__ void k_t_cl_lc(const float* __restrict__ in, f16* __restrict__ out,
                          int Cc, int Ld, long long total) {
  long long i = (long long)blockIdx.x * blockDim.x + threadIdx.x;
  if (i >= total) return;
  long long cl = (long long)Cc * Ld;
  int n = (int)(i / cl);
  long long rem = i - (long long)n * cl;
  int c = (int)(rem / Ld);
  int l = (int)(rem - (long long)c * Ld);
  out[(long long)n * cl + (long long)l * Cc + c] = (f16)in[i];
}

// (N,L,C) f16 -> (N,C,L) f16
__global__ void k_t_lc_cl_h(const f16* __restrict__ in, f16* __restrict__ out,
                            int Cc, int Ld, long long total) {
  long long i = (long long)blockIdx.x * blockDim.x + threadIdx.x;
  if (i >= total) return;
  long long cl = (long long)Cc * Ld;
  int n = (int)(i / cl);
  long long rem = i - (long long)n * cl;
  int l = (int)(rem / Cc);
  int c = (int)(rem - (long long)l * Cc);
  out[(long long)n * cl + (long long)c * Ld + l] = in[i];
}

// (N,L,C) f32 -> (N,C,L) f32  (final output)
__global__ void k_final_t(const float* __restrict__ in, float* __restrict__ out,
                          int Cc, int Ld, long long total) {
  long long i = (long long)blockIdx.x * blockDim.x + threadIdx.x;
  if (i >= total) return;
  long long cl = (long long)Cc * Ld;
  int n = (int)(i / cl);
  long long rem = i - (long long)n * cl;
  int l = (int)(rem / Cc);
  int c = (int)(rem - (long long)l * Cc);
  out[(long long)n * cl + (long long)c * Ld + l] = in[i];
}

// elementwise BN (channel = low 8 bits of index, C=256) -> f32 + f16
__global__ void k_bnelem(const float* __restrict__ in, const float* __restrict__ s,
                         const float* __restrict__ t, float* __restrict__ oF,
                         f16* __restrict__ oH, long long total) {
  long long i = (long long)blockIdx.x * blockDim.x + threadIdx.x;
  if (i >= total) return;
  int c = (int)(i & 255);
  float v = in[i] * s[c] + t[c];
  oF[i] = v;
  oH[i] = (f16)v;
}

// z = relu(instnorm(up + h3)) over L per (n,c); layout (N,L,C), in-place in h3
__global__ void k_resnorm(const float* __restrict__ up, float* __restrict__ h3,
                          int Cc, int Ld) {
  int c = blockIdx.x, n = blockIdx.y, tid = threadIdx.x;
  const float* pu = up + (long long)n * Ld * Cc + c;
  float* ph = h3 + (long long)n * Ld * Cc + c;
  __shared__ float sA[256], sB[256];
  float s = 0.f, s2 = 0.f;
  for (int l = tid; l < Ld; l += 256) {
    float v = pu[(long long)l * Cc] + ph[(long long)l * Cc];
    s += v; s2 += v * v;
  }
  sA[tid] = s; sB[tid] = s2; __syncthreads();
  for (int st = 128; st > 0; st >>= 1) {
    if (tid < st) { sA[tid] += sA[tid + st]; sB[tid] += sB[tid + st]; }
    __syncthreads();
  }
  float mean = sA[0] / Ld;
  float var = sB[0] / Ld - mean * mean;
  float rinv = rsqrtf(var + 1e-5f);
  for (int l = tid; l < Ld; l += 256) {
    float v = pu[(long long)l * Cc] + ph[(long long)l * Cc];
    v = (v - mean) * rinv;
    ph[(long long)l * Cc] = v > 0.f ? v : 0.f;
  }
}

// row softmax over Lk, f32 scores -> f16 probs
__global__ void k_softmax(const float* __restrict__ sc, f16* __restrict__ pr, int Lk) {
  int q = blockIdx.x, tid = threadIdx.x;
  const float* row = sc + (long long)q * Lk;
  f16* prow = pr + (long long)q * Lk;
  __shared__ float sm[256];
  float m = -1e30f;
  for (int i = tid; i < Lk; i += 256) m = fmaxf(m, row[i]);
  sm[tid] = m; __syncthreads();
  for (int st = 128; st > 0; st >>= 1) {
    if (tid < st) sm[tid] = fmaxf(sm[tid], sm[tid + st]);
    __syncthreads();
  }
  m = sm[0]; __syncthreads();
  float sum = 0.f;
  for (int i = tid; i < Lk; i += 256) sum += __expf(row[i] - m);
  sm[tid] = sum; __syncthreads();
  for (int st = 128; st > 0; st >>= 1) {
    if (tid < st) sm[tid] += sm[tid + st];
    __syncthreads();
  }
  float rinv = 1.f / sm[0];
  for (int i = tid; i < Lk; i += 256) prow[i] = (f16)(__expf(row[i] - m) * rinv);
}

// ---------------------------------------------------------------------------
extern "C" void kernel_launch(void* const* d_in, const int* in_sizes, int n_in,
                              void* d_out, int out_size, void* d_ws, size_t ws_size,
                              hipStream_t stream) {
  (void)in_sizes; (void)n_in; (void)out_size; (void)ws_size;
  const int N = 4, C = 256, M = 512, Lx = 1024, L = 2048, Ly = 2048, H = 8, DH = 32;
  typedef long long ll;

  const float* x    = (const float*)d_in[0];
  const float* y    = (const float*)d_in[1];
  const float* upw  = (const float*)d_in[2];
  const float* upb  = (const float*)d_in[3];
  const float* c1w  = (const float*)d_in[4];
  const float* c1b  = (const float*)d_in[5];
  const float* bn1g = (const float*)d_in[6],  *bn1b = (const float*)d_in[7];
  const float* bn1m = (const float*)d_in[8],  *bn1v = (const float*)d_in[9];
  const float* c2w  = (const float*)d_in[10];
  const float* c2b  = (const float*)d_in[11];
  const float* bn2g = (const float*)d_in[12], *bn2b = (const float*)d_in[13];
  const float* bn2m = (const float*)d_in[14], *bn2v = (const float*)d_in[15];
  const float* c3w  = (const float*)d_in[16];
  const float* c3b  = (const float*)d_in[17];
  const float* bn3g = (const float*)d_in[18], *bn3b = (const float*)d_in[19];
  const float* bn3m = (const float*)d_in[20], *bn3v = (const float*)d_in[21];
  const float* aw1[4] = {(const float*)d_in[22], (const float*)d_in[23],
                         (const float*)d_in[24], (const float*)d_in[25]};
  const float* n1g = (const float*)d_in[26], *n1b = (const float*)d_in[27];
  const float* n1m = (const float*)d_in[28], *n1v = (const float*)d_in[29];
  const float* aw2[4] = {(const float*)d_in[30], (const float*)d_in[31],
                         (const float*)d_in[32], (const float*)d_in[33]};
  const float* n2g = (const float*)d_in[34], *n2b = (const float*)d_in[35];
  const float* n2m = (const float*)d_in[36], *n2v = (const float*)d_in[37];

  // ---- workspace layout (≈90 MB, buffers reused across phases) ----
  char* base = (char*)d_ws;
  size_t off = 0;
  auto alloc = [&](size_t bytes) -> void* {
    void* p = base + off;
    off += (bytes + 255) & ~(size_t)255;
    return p;
  };
  f16* wUp = (f16*)alloc(3LL * C * C * sizeof(f16));
  f16* wC1 = (f16*)alloc(3LL * M * C * sizeof(f16));
  f16* wC2 = (f16*)alloc(3LL * M * M * sizeof(f16));
  f16* wC3 = (f16*)alloc(3LL * C * M * sizeof(f16));
  f16* wA[8];
  for (int i = 0; i < 8; ++i) wA[i] = (f16*)alloc((size_t)C * C * sizeof(f16));
  f16* zrow  = (f16*)alloc((size_t)L * sizeof(f16));           // zeroed K-row
  f16* xt16  = (f16*)alloc((size_t)N * Lx * C * sizeof(f16));
  f16* yt16  = (f16*)alloc((size_t)N * Ly * C * sizeof(f16));
  f16* bufU  = (f16*)alloc((size_t)N * L * C * sizeof(f16));   // up16 -> a1_16 -> a2_16
  f16* bufB1 = (f16*)alloc((size_t)N * L * M * sizeof(f16));   // h1 -> (q16,k16)
  f16* bufB2 = (f16*)alloc((size_t)N * L * M * sizeof(f16));   // h2 -> (v16,vt16)
  f16* bufO16 = (f16*)alloc((size_t)N * L * C * sizeof(f16));  // attnO f16
  f16* probs = (f16*)alloc((size_t)L * L * sizeof(f16));
  float* st = (float*)alloc(12 * 512 * sizeof(float));
  float* f0 = (float*)alloc((size_t)N * L * C * sizeof(float)); // up -> x2
  float* fz = (float*)alloc((size_t)N * L * C * sizeof(float)); // h3/z -> final
  float* f1 = (float*)alloc((size_t)N * L * C * sizeof(float)); // a1 -> a2
  float* f2 = (float*)alloc((size_t)N * L * C * sizeof(float)); // attnO f32
  float* fs = (float*)alloc((size_t)L * L * sizeof(float));     // scores slice

  float *sUp = st, *tUp = st + 512, *s1 = st + 1024, *t1 = st + 1536,
        *s2 = st + 2048, *t2 = st + 2560, *s3 = st + 3072, *t3 = st + 3584,
        *sn1 = st + 4096, *tn1 = st + 4608, *sn2 = st + 5120, *tn2 = st + 5632;

  auto g1 = [](ll total) { return dim3((unsigned)((total + 255) / 256)); };
  auto gemm = [&](const f16* A, int lda, ll sAn, const f16* BT, int ldb, ll sBn,
                  float* oF, f16* oH, int ldo, ll sOn, int cob,
                  const float* sc, const float* sh, float us,
                  const float* res, ll sRn,
                  int Mm, int Kk, int Lo, int Li, int nb, int nt, int mode, int relu) {
    int waves = (Mm / 32) * (Lo / 64) * nb;
    int blocks = (waves + 3) / 4;
    k_gemm<<<blocks, 128, 0, stream>>>(A, lda, sAn, BT, ldb, sBn, zrow, oF, oH,
                                       ldo, sOn, cob, sc, sh, us, res, sRn,
                                       Mm, Kk, Lo, Li, nb, nt, mode, relu);
  };

  // ---- prep: zero-row, BN folds, weight conversion, input transposes ----
  k_zero_h<<<g1(L), 256, 0, stream>>>(zrow, L);
  k_upprep<<<1, 256, 0, stream>>>(upb, sUp, tUp, C);
  k_bnprep<<<2, 256, 0, stream>>>(bn1g, bn1b, bn1m, bn1v, c1b, s1, t1, M);
  k_bnprep<<<2, 256, 0, stream>>>(bn2g, bn2b, bn2m, bn2v, c2b, s2, t2, M);
  k_bnprep<<<1, 256, 0, stream>>>(bn3g, bn3b, bn3m, bn3v, c3b, s3, t3, C);
  k_bnprep<<<1, 256, 0, stream>>>(n1g, n1b, n1m, n1v, nullptr, sn1, tn1, C);
  k_bnprep<<<1, 256, 0, stream>>>(n2g, n2b, n2m, n2v, nullptr, sn2, tn2, C);

  k_cvt_w3<<<g1(3LL * C * C), 256, 0, stream>>>(upw, wUp, C, C, 3, C * 3); // (I,O,3)
  k_cvt_w3<<<g1(3LL * M * C), 256, 0, stream>>>(c1w, wC1, M, C, C * 3, 3); // (O,I,3)
  k_cvt_w3<<<g1(3LL * M * M), 256, 0, stream>>>(c2w, wC2, M, M, M * 3, 3);
  k_cvt_w3<<<g1(3LL * C * M), 256, 0, stream>>>(c3w, wC3, C, M, M * 3, 3);
  for (int i = 0; i < 4; ++i) {
    k_cvt_mat<<<g1((ll)C * C), 256, 0, stream>>>(aw1[i], wA[i], (ll)C * C);
    k_cvt_mat<<<g1((ll)C * C), 256, 0, stream>>>(aw2[i], wA[4 + i], (ll)C * C);
  }
  k_t_cl_lc<<<g1((ll)N * C * Lx), 256, 0, stream>>>(x, xt16, C, Lx, (ll)N * C * Lx);
  k_t_cl_lc<<<g1((ll)N * C * Ly), 256, 0, stream>>>(y, yt16, C, Ly, (ll)N * C * Ly);

  // ---- upsample (ConvTranspose1d k3 s2 p1 op1) -> f0 (f32) + bufU (f16) ----
  gemm(wUp, C, 0, xt16, C, (ll)Lx * C, f0, bufU, C, (ll)L * C, 0,
       sUp, tUp, 1.f, nullptr, 0, C, C, L, Lx, N, 3, 1, 0);

  // ---- resblock convs (BN+bias folded, ReLU fused) ----
  gemm(wC1, C, 0, bufU, C, (ll)L * C, nullptr, bufB1, M, (ll)L * M, 0,
       s1, t1, 1.f, nullptr, 0, M, C, L, L, N, 3, 0, 1);
  gemm(wC2, M, 0, bufB1, M, (ll)L * M, nullptr, bufB2, M, (ll)L * M, 0,
       s2, t2, 1.f, nullptr, 0, M, M, L, L, N, 3, 0, 1);
  gemm(wC3, M, 0, bufB2, M, (ll)L * M, fz, nullptr, C, (ll)L * C, 0,
       s3, t3, 1.f, nullptr, 0, C, M, L, L, N, 3, 0, 0);

  // z = relu(instnorm(up + h3)), in-place in fz
  k_resnorm<<<dim3(C, N), 256, 0, stream>>>(f0, fz, C, L);

  // ---- self-attention block ----
  k_bnelem<<<g1((ll)N * L * C), 256, 0, stream>>>(fz, sn1, tn1, f1, bufU, (ll)N * L * C);
  f16* q16 = bufB1;
  f16* k16 = bufB1 + (size_t)N * L * C;
  f16* v16 = bufB2;
  f16* vt16 = bufB2 + (size_t)N * L * C;

  // projections (self): q,k,v from a1
  gemm(wA[0], C, 0, bufU, C, (ll)L * C, nullptr, q16, C, (ll)L * C, 0,
       nullptr, nullptr, 1.f, nullptr, 0, C, C, L, L, N, 1, 0, 0);
  gemm(wA[1], C, 0, bufU, C, (ll)L * C, nullptr, k16, C, (ll)L * C, 0,
       nullptr, nullptr, 1.f, nullptr, 0, C, C, L, L, N, 1, 0, 0);
  gemm(wA[2], C, 0, bufU, C, (ll)L * C, nullptr, v16, C, (ll)L * C, 0,
       nullptr, nullptr, 1.f, nullptr, 0, C, C, L, L, N, 1, 0, 0);
  k_t_lc_cl_h<<<g1((ll)N * L * C), 256, 0, stream>>>(v16, vt16, C, L, (ll)N * L * C);

  auto attn_core = [&]() {
    for (int n = 0; n < N; ++n) {
      for (int h = 0; h < H; ++h) {
        const f16* qs = q16 + (ll)n * L * C + h * DH;
        const f16* ks = k16 + (ll)n * L * C + h * DH;
        const f16* vs = vt16 + (ll)n * C * L + (ll)h * DH * L;
        // scores[q*Lk + k] = (K_h Q_h^T)/16
        gemm(ks, C, 0, qs, C, 0, fs, nullptr, L, 0, 0,
             nullptr, nullptr, 0.0625f, nullptr, 0, L, DH, L, L, 1, 1, 0, 0);
        k_softmax<<<L, 256, 0, stream>>>(fs, probs, L);
        // out[q, h*32+d] = V_h^T probs^T
        gemm(vs, L, 0, probs, L, 0, f2 + (ll)n * L * C, bufO16 + (ll)n * L * C,
             C, 0, h * DH, nullptr, nullptr, 1.f, nullptr, 0,
             DH, L, L, L, 1, 1, 0, 0);
      }
    }
  };
  attn_core();

  // o-projection + residual: x2 = a1 + attnO @ wo^T  -> f0
  gemm(wA[3], C, 0, bufO16, C, (ll)L * C, f0, nullptr, C, (ll)L * C, 0,
       nullptr, nullptr, 1.f, f1, (ll)L * C, C, C, L, L, N, 1, 0, 0);

  // ---- cross-attention block ----
  k_bnelem<<<g1((ll)N * L * C), 256, 0, stream>>>(f0, sn2, tn2, f1, bufU, (ll)N * L * C);
  gemm(wA[4], C, 0, bufU, C, (ll)L * C, nullptr, q16, C, (ll)L * C, 0,
       nullptr, nullptr, 1.f, nullptr, 0, C, C, L, L, N, 1, 0, 0);
  gemm(wA[5], C, 0, yt16, C, (ll)Ly * C, nullptr, k16, C, (ll)L * C, 0,
       nullptr, nullptr, 1.f, nullptr, 0, C, C, Ly, Ly, N, 1, 0, 0);
  gemm(wA[6], C, 0, yt16, C, (ll)Ly * C, nullptr, v16, C, (ll)L * C, 0,
       nullptr, nullptr, 1.f, nullptr, 0, C, C, Ly, Ly, N, 1, 0, 0);
  k_t_lc_cl_h<<<g1((ll)N * L * C), 256, 0, stream>>>(v16, vt16, C, L, (ll)N * L * C);
  attn_core();

  // o-projection + residual -> fz, then transpose to (N,C,L) output
  gemm(wA[7], C, 0, bufO16, C, (ll)L * C, fz, nullptr, C, (ll)L * C, 0,
       nullptr, nullptr, 1.f, f1, (ll)L * C, C, C, L, L, N, 1, 0, 0);
  k_final_t<<<g1((ll)N * L * C), 256, 0, stream>>>(fz, (float*)d_out, C, L, (ll)N * L * C);
}
